// Transformer_61753039782189
// MI455X (gfx1250) — compile-verified, matrix operand order checked by reading, not run on hardware
//
#include <hip/hip_runtime.h>
#include <math.h>

// Problem constants (match reference)
#define BB 4
#define TT 1024
#define CC 768
#define HH 12
#define SS 64          // head dim = C/H
#define DD 4           // layers
#define GG 4
#define FFN (GG*CC)    // 3072
#define VV 32000
#define BT (BB*TT)     // 4096 rows

typedef __attribute__((ext_vector_type(16))) _Float16 v16h;
typedef __attribute__((ext_vector_type(8)))  float    v8f;
typedef __attribute__((ext_vector_type(4)))  unsigned int u32x4;
typedef __attribute__((ext_vector_type(8)))  int i32x8;
typedef __attribute__((ext_vector_type(4)))  int i32x4;

// ---------------------------------------------------------------------------
// Embedding: h[b,t,c] = wordemb[x[b,t],c] + posemb[t,c]
// ---------------------------------------------------------------------------
__global__ __launch_bounds__(256)
void embed_kernel(const int* __restrict__ x, const float* __restrict__ wordemb,
                  const float* __restrict__ posemb, float* __restrict__ h)
{
    int idx = blockIdx.x * 256 + threadIdx.x;       // over BT*CC
    if (idx >= BT * CC) return;
    int c  = idx % CC;
    int bt = idx / CC;
    int t  = bt % TT;
    int tok = x[bt];
    h[idx] = wordemb[(size_t)tok * CC + c] + posemb[(size_t)t * CC + c];
}

// ---------------------------------------------------------------------------
// Tiled WMMA GEMM with TDM-staged B tiles:
//   out[M,N] = A[M,K] * W[K,N] (+bias) (optional relu)
//
//  - block = 256 threads (8 waves), block tile 64(M) x 128(N), K-step 32
//  - B tile (32x128 f32) is DMA'd into LDS by the Tensor Data Mover
//    (tensor_load_to_lds, TENSORcnt), issued by wave 0; descriptor per
//    CDNA5 ISA (2D, data_size=4B)
//  - A tile (64x32) converted f32->f16 directly into LDS (global_load_b128)
//  - after s_wait_tensorcnt + barrier, all threads convert the f32 B tile
//    to f16 *transposed* (Bt[n][k]) so fragments are contiguous 16B groups
//  - each wave owns a 32x32 macro-tile: 2x2 fragments -> 4 WMMAs per K-step
//  - fragment layout per CDNA5 ISA 7.12.2: half h of lane L maps to
//      K = (L>=16 ? 8 : 0) + h + (h>=8 ? 8 : 0)  -> 2x ds_load_b128 per frag
//
// Requires: M%64==0, N%128==0, K%32==0 (true for all shapes used here).
// ---------------------------------------------------------------------------
#define LDA 40   // padded LDS stride (halves) for A rows
#define LDB 40   // padded LDS stride (halves) for Bt rows

__device__ inline uint4 cvt8_f16(const float4 a, const float4 b)
{
    union { _Float16 h[8]; uint4 u; } p;
    p.h[0] = (_Float16)a.x; p.h[1] = (_Float16)a.y;
    p.h[2] = (_Float16)a.z; p.h[3] = (_Float16)a.w;
    p.h[4] = (_Float16)b.x; p.h[5] = (_Float16)b.y;
    p.h[6] = (_Float16)b.z; p.h[7] = (_Float16)b.w;
    return p.u;
}

__device__ inline v16h load_frag(const _Float16* __restrict__ rowp, int koff)
{
    v16h f;
    uint4* fp = reinterpret_cast<uint4*>(&f);
    fp[0] = *reinterpret_cast<const uint4*>(rowp + koff);        // K group lo
    fp[1] = *reinterpret_cast<const uint4*>(rowp + koff + 16);   // K group hi
    return f;
}

__global__ __launch_bounds__(256)
void wmma_gemm_kernel(const float* __restrict__ A, const float* __restrict__ W,
                      const float* __restrict__ bias, float* __restrict__ out,
                      int M, int K, int N, int relu)
{
    __shared__ _Float16 As[64 * LDA];       // [row][k] f16      5120 B
    __shared__ _Float16 Bt[128 * LDB];      // [col][k] f16     10240 B
    __shared__ float    Bs32[32 * 128];     // TDM B tile f32   16384 B

    const int tid    = threadIdx.x;
    const int lane   = tid & 31;
    const int lane15 = lane & 15;
    const int koff   = (lane >> 4) * 8;   // halves: 0 (lanes 0-15) / 8 (16-31)
    const int wave   = tid >> 5;
    const int wmb    = (wave & 1) * 32;   // wave M offset in block tile
    const int wnb    = (wave >> 1) * 32;  // wave N offset in block tile

    const int m0 = blockIdx.x * 64;
    const int n0 = blockIdx.y * 128;

    // Staging assignments
    const int arow = tid >> 2;            // 0..63   (4 threads per A row)
    const int kseg = (tid & 3) * 8;       // 0,8,16,24
    const int bk   = tid >> 3;            // 0..31   (8 threads per B k-row)
    const int nseg = (tid & 7) * 16;      // 0..112

    const float* __restrict__ agp = A + (size_t)(m0 + arow) * K + kseg;
    const unsigned lds_b32 = (unsigned)(uintptr_t)(&Bs32[0]);   // LDS byte offset

    v8f acc[2][2] = {};

    for (int k0 = 0; k0 < K; k0 += 32) {
        __syncthreads();   // previous compute done reading As/Bt; Bs32 consumed

        // ---- TDM: DMA the 32x128 f32 B tile (rows k0..k0+31, cols n0..n0+127)
        if (wave == 0) {
            const unsigned long long ga =
                (unsigned long long)(uintptr_t)(W + (size_t)k0 * N + n0);
            u32x4 g0 = { 1u,                                   // count=1
                         lds_b32,                              // lds_addr
                         (unsigned)(ga & 0xffffffffu),
                         (unsigned)((ga >> 32) & 0x01ffffffu) | (2u << 30) }; // type=2
            i32x8 g1 = { (int)(2u << 16),                                  // data_size=4B
                         (int)(((unsigned)N & 0xffffu) << 16),             // dim0 lo16
                         (int)((((unsigned)N >> 16) & 0xffffu) | (32u << 16)), // dim0 hi / dim1=32
                         (int)(128u << 16),                                // tile_dim0=128
                         (int)32u,                                         // tile_dim1=32
                         (int)(unsigned)N,                                 // dim0_stride lo32
                         0, 0 };
            i32x4 gz4 = { 0, 0, 0, 0 };
            i32x8 gz8 = { 0, 0, 0, 0, 0, 0, 0, 0 };
            __builtin_amdgcn_tensor_load_to_lds(g0, g1, gz4, gz4, gz8, 0);
        }

        // ---- stage A tile: 8 consecutive floats -> 8 halves (16B store)
        {
            const float4* ap = (const float4*)(agp + k0);
            float4 a0 = ap[0], a1 = ap[1];
            *(uint4*)&As[arow * LDA + kseg] = cvt8_f16(a0, a1);
        }
        if (k0 + 32 < K)
            __builtin_prefetch((const void*)(agp + k0 + 32), 0, 3);

        if (wave == 0) __builtin_amdgcn_s_wait_tensorcnt(0);
        __syncthreads();   // Bs32 (TDM) and As visible to all waves

        // ---- convert B tile f32 -> f16 transposed: Bt[n][k]
        {
            const float4* wp = (const float4*)&Bs32[bk * 128 + nseg];
#pragma unroll
            for (int j = 0; j < 4; ++j) {
                float4 wv = wp[j];
                _Float16* d = &Bt[(nseg + j * 4) * LDB + bk];
                d[0 * LDB] = (_Float16)wv.x;
                d[1 * LDB] = (_Float16)wv.y;
                d[2 * LDB] = (_Float16)wv.z;
                d[3 * LDB] = (_Float16)wv.w;
            }
        }
        __syncthreads();   // Bt ready

        // ---- fragments + 2x2 WMMA
        v16h af[2], bf[2];
#pragma unroll
        for (int mi = 0; mi < 2; ++mi)
            af[mi] = load_frag(&As[(wmb + mi * 16 + lane15) * LDA], koff);
#pragma unroll
        for (int ni = 0; ni < 2; ++ni)
            bf[ni] = load_frag(&Bt[(wnb + ni * 16 + lane15) * LDB], koff);

#pragma unroll
        for (int mi = 0; mi < 2; ++mi)
#pragma unroll
            for (int ni = 0; ni < 2; ++ni)
                acc[mi][ni] = __builtin_amdgcn_wmma_f32_16x16x32_f16(
                    false, af[mi], false, bf[ni],
                    (short)0, acc[mi][ni], false, false);
    }

    // ---- epilogue
#pragma unroll
    for (int mi = 0; mi < 2; ++mi) {
        const int mbase = m0 + wmb + mi * 16 + (lane >> 4) * 8;
#pragma unroll
        for (int ni = 0; ni < 2; ++ni) {
            const int col = n0 + wnb + ni * 16 + lane15;
            const float bv = bias ? bias[col] : 0.0f;
#pragma unroll
            for (int r = 0; r < 8; ++r) {
                float v = acc[mi][ni][r] + bv;
                if (relu) v = v > 0.0f ? v : 0.0f;
                out[(size_t)(mbase + r) * N + col] = v;
            }
        }
    }
}

// ---------------------------------------------------------------------------
// Causal attention, online softmax. One thread per (b, head, q-position).
// Scale by 1/sqrt(C) (reference scales by sqrt(channels), not head_dim).
// ---------------------------------------------------------------------------
__global__ __launch_bounds__(64)
void attention_kernel(const float* __restrict__ q, const float* __restrict__ k,
                      const float* __restrict__ v, float* __restrict__ att)
{
    int gid = blockIdx.x * 64 + threadIdx.x;    // over BB*HH*TT
    if (gid >= BB * HH * TT) return;
    const int qpos = gid % TT;
    const int hd   = (gid / TT) % HH;
    const int b    = gid / (TT * HH);

    const float scale = rsqrtf((float)CC);
    const size_t rowbase = ((size_t)(b * TT + qpos)) * CC + hd * SS;

    float qr[SS];
#pragma unroll
    for (int s = 0; s < SS; ++s) qr[s] = q[rowbase + s] * scale;

    const float NEG_INF = -__builtin_huge_valf();
    float m = NEG_INF, l = 0.0f;
    float o[SS];
#pragma unroll
    for (int s = 0; s < SS; ++s) o[s] = 0.0f;

    for (int kp = 0; kp <= qpos; ++kp) {
        const size_t kb = ((size_t)(b * TT + kp)) * CC + hd * SS;
        float dot = 0.0f;
#pragma unroll
        for (int s = 0; s < SS; ++s) dot += qr[s] * k[kb + s];
        const float nm = fmaxf(m, dot);
        const float sc = __expf(m - nm);     // 0 on first iter (m = -inf)
        const float p  = __expf(dot - nm);
        l = l * sc + p;
#pragma unroll
        for (int s = 0; s < SS; ++s) o[s] = o[s] * sc + p * v[kb + s];
        m = nm;
    }
    const float inv = 1.0f / l;
#pragma unroll
    for (int s = 0; s < SS; ++s) att[rowbase + s] = o[s] * inv;
}

// ---------------------------------------------------------------------------
// h_out = h_in + LayerNorm(delta)*g + beta   (one block per (b,t) row)
// ---------------------------------------------------------------------------
__global__ __launch_bounds__(256)
void ln_residual_kernel(const float* __restrict__ hin, const float* __restrict__ delta,
                        const float* __restrict__ g, const float* __restrict__ be,
                        float* __restrict__ hout)
{
    const int rowi = blockIdx.x;
    const float* x = delta + (size_t)rowi * CC;
    __shared__ float r1[256], r2[256];

    float s = 0.0f, ss = 0.0f;
    for (int c = threadIdx.x; c < CC; c += 256) { float t = x[c]; s += t; ss += t * t; }
    r1[threadIdx.x] = s; r2[threadIdx.x] = ss;
    __syncthreads();
    for (int off = 128; off > 0; off >>= 1) {
        if (threadIdx.x < off) {
            r1[threadIdx.x] += r1[threadIdx.x + off];
            r2[threadIdx.x] += r2[threadIdx.x + off];
        }
        __syncthreads();
    }
    const float mu   = r1[0] * (1.0f / CC);
    const float var  = r2[0] * (1.0f / CC) - mu * mu;
    const float rstd = rsqrtf(var + 1e-5f);

    for (int c = threadIdx.x; c < CC; c += 256) {
        const size_t i = (size_t)rowi * CC + c;
        hout[i] = hin[i] + (x[c] - mu) * rstd * g[c] + be[c];
    }
}

// ---------------------------------------------------------------------------
// Loss: mean over rows of (logsumexp(logits_row) - logits_row[y])
// ---------------------------------------------------------------------------
__global__ void zero_loss_kernel(float* loss) { *loss = 0.0f; }

__global__ __launch_bounds__(256)
void loss_kernel(const float* __restrict__ logits, const int* __restrict__ y,
                 float* __restrict__ loss)
{
    const int rowi = blockIdx.x;                 // BT rows
    const float* lr = logits + (size_t)rowi * VV;
    __shared__ float red[256];

    float mx = -__builtin_huge_valf();
    for (int c = threadIdx.x; c < VV; c += 256) mx = fmaxf(mx, lr[c]);
    red[threadIdx.x] = mx;
    __syncthreads();
    for (int off = 128; off > 0; off >>= 1) {
        if (threadIdx.x < off) red[threadIdx.x] = fmaxf(red[threadIdx.x], red[threadIdx.x + off]);
        __syncthreads();
    }
    mx = red[0];
    __syncthreads();

    float sum = 0.0f;
    for (int c = threadIdx.x; c < VV; c += 256) sum += __expf(lr[c] - mx);
    red[threadIdx.x] = sum;
    __syncthreads();
    for (int off = 128; off > 0; off >>= 1) {
        if (threadIdx.x < off) red[threadIdx.x] += red[threadIdx.x + off];
        __syncthreads();
    }
    if (threadIdx.x == 0) {
        const float lse = mx + logf(red[0]);
        const float nll = lse - lr[y[rowi]];
        atomicAdd(loss, nll * (1.0f / (float)BT));
    }
}

// ---------------------------------------------------------------------------
// Orchestration
// ---------------------------------------------------------------------------
extern "C" void kernel_launch(void* const* d_in, const int* in_sizes, int n_in,
                              void* d_out, int out_size, void* d_ws, size_t ws_size,
                              hipStream_t stream)
{
    (void)in_sizes; (void)n_in; (void)out_size; (void)ws_size;

    const int*   x       = (const int*)  d_in[0];
    const int*   y       = (const int*)  d_in[1];
    const float* wordemb = (const float*)d_in[2];
    const float* posemb  = (const float*)d_in[3];
    const float* Wq      = (const float*)d_in[4];
    const float* Wk      = (const float*)d_in[5];
    const float* Wv      = (const float*)d_in[6];
    const float* Wo      = (const float*)d_in[7];
    const float* bo      = (const float*)d_in[8];
    const float* ln1g    = (const float*)d_in[9];
    const float* ln1b    = (const float*)d_in[10];
    const float* W1      = (const float*)d_in[11];
    const float* b1      = (const float*)d_in[12];
    const float* W2      = (const float*)d_in[13];
    const float* b2      = (const float*)d_in[14];
    const float* ln2g    = (const float*)d_in[15];
    const float* ln2b    = (const float*)d_in[16];
    const float* Wl      = (const float*)d_in[17];
    const float* bl      = (const float*)d_in[18];

    const size_t BTC = (size_t)BT * CC;           // 3.146M floats
    float* ws   = (float*)d_ws;
    float* h    = ws;                              // BTC
    float* tmp  = h    + BTC;                      // BTC
    float* qb   = tmp  + BTC;                      // BTC
    float* kb   = qb   + BTC;                      // BTC
    float* vb   = kb   + BTC;                      // BTC
    float* attb = vb   + BTC;                      // BTC
    float* ffb  = attb + BTC;                      // BT*FFN

    float* logits = (float*)d_out;                 // BT*VV
    float* loss   = logits + (size_t)BT * VV;      // 1

    // Embedding
    embed_kernel<<<(BT * CC + 255) / 256, 256, 0, stream>>>(x, wordemb, posemb, h);

    const dim3 gC (BT / 64, CC  / 128);   // (64, 6)
    const dim3 gF (BT / 64, FFN / 128);   // (64, 24)
    const dim3 gV (BT / 64, VV  / 128);   // (64, 250)

    for (int l = 0; l < DD; ++l) {
        const float* wq = Wq + (size_t)l * CC * CC;
        const float* wk = Wk + (size_t)l * CC * CC;
        const float* wv = Wv + (size_t)l * CC * CC;
        const float* wo = Wo + (size_t)l * CC * CC;
        const float* w1 = W1 + (size_t)l * CC * FFN;
        const float* w2 = W2 + (size_t)l * FFN * CC;

        wmma_gemm_kernel<<<gC, 256, 0, stream>>>(h, wq, nullptr, qb, BT, CC, CC, 0);
        wmma_gemm_kernel<<<gC, 256, 0, stream>>>(h, wk, nullptr, kb, BT, CC, CC, 0);
        wmma_gemm_kernel<<<gC, 256, 0, stream>>>(h, wv, nullptr, vb, BT, CC, CC, 0);

        attention_kernel<<<(BB * HH * TT + 63) / 64, 64, 0, stream>>>(qb, kb, vb, attb);

        wmma_gemm_kernel<<<gC, 256, 0, stream>>>(attb, wo, bo + (size_t)l * CC, tmp,
                                                 BT, CC, CC, 0);
        ln_residual_kernel<<<BT, 256, 0, stream>>>(h, tmp, ln1g + (size_t)l * CC,
                                                   ln1b + (size_t)l * CC, h);

        wmma_gemm_kernel<<<gF, 256, 0, stream>>>(h, w1, b1 + (size_t)l * FFN, ffb,
                                                 BT, CC, FFN, 1);
        wmma_gemm_kernel<<<gC, 256, 0, stream>>>(ffb, w2, b2 + (size_t)l * CC, tmp,
                                                 BT, FFN, CC, 0);
        ln_residual_kernel<<<BT, 256, 0, stream>>>(h, tmp, ln2g + (size_t)l * CC,
                                                   ln2b + (size_t)l * CC, h);
    }

    // Final logits + loss
    wmma_gemm_kernel<<<gV, 256, 0, stream>>>(h, Wl, bl, logits, BT, CC, VV, 0);
    zero_loss_kernel<<<1, 1, 0, stream>>>(loss);
    loss_kernel<<<BT, 256, 0, stream>>>(logits, y, loss);
}
